// AttentiveMatch_38268158608099
// MI455X (gfx1250) — compile-verified, matrix operand order checked by reading, not run on hardware
//
#include <hip/hip_runtime.h>
#include <hip/hip_bf16.h>

typedef __attribute__((ext_vector_type(16))) _Float16 v16h;
typedef __attribute__((ext_vector_type(8)))  float    v8f;

#define B_ 64
#define S_ 512
#define D_ 768

#define BM 128
#define BN 256
#define BK 32
#define LDT 40   // LDS row stride in halfs: 80B row -> 16B aligned, conflict-free

// ---------------------------------------------------------------------------
// CDNA5 async global->LDS copy (ASYNCcnt path), with sync fallback.
// Probe-verified prototype: (v4i addrspace(1)*, v4i addrspace(3)*, imm, imm)
// ---------------------------------------------------------------------------
#if __has_builtin(__builtin_amdgcn_global_load_async_to_lds_b128)
#define HAVE_ASYNC_LDS 1
#else
#define HAVE_ASYNC_LDS 0
#endif

typedef int i32x4_v __attribute__((vector_size(16)));
typedef __attribute__((address_space(1))) i32x4_v* as1_b128_t;
typedef __attribute__((address_space(3))) i32x4_v* as3_b128_t;

__device__ __forceinline__ void copy16_g2l(const _Float16* g, _Float16* l) {
#if HAVE_ASYNC_LDS
  __builtin_amdgcn_global_load_async_to_lds_b128(
      (as1_b128_t)(g), (as3_b128_t)(l), 0, 0);
#else
  *(uint4*)l = *(const uint4*)g;
#endif
}

__device__ __forceinline__ void async_drain() {
#if HAVE_ASYNC_LDS
#if __has_builtin(__builtin_amdgcn_s_wait_asynccnt)
  __builtin_amdgcn_s_wait_asynccnt(0);
#else
  asm volatile("s_wait_asynccnt 0x0" ::: "memory");
#endif
#endif
}

// ---------------------------------------------------------------------------
// Kernel 1: L2-normalize p and q rows to f16, plus transposed raw q (f16).
// One block per (b,s) row; 768 = 3 x 256 elements per thread.
// ---------------------------------------------------------------------------
__global__ __launch_bounds__(256) void prep_kernel(
    const float* __restrict__ p, const float* __restrict__ q,
    _Float16* __restrict__ pn, _Float16* __restrict__ qn,
    _Float16* __restrict__ qT)
{
  const int row = blockIdx.x;            // b*S + s
  const int b = row >> 9, s = row & 511;
  const int tid = threadIdx.x;
  const float* pr = p + (long)row * D_;
  const float* qr = q + (long)row * D_;

  float pv[3], qv[3];
  float ssp = 0.f, ssq = 0.f;
#pragma unroll
  for (int i = 0; i < 3; ++i) {
    float x = pr[tid + i * 256]; pv[i] = x; ssp += x * x;
    float y = qr[tid + i * 256]; qv[i] = y; ssq += y * y;
  }
  __shared__ float red[2][256];
  red[0][tid] = ssp; red[1][tid] = ssq;
  __syncthreads();
  for (int off = 128; off > 0; off >>= 1) {
    if (tid < off) {
      red[0][tid] += red[0][tid + off];
      red[1][tid] += red[1][tid + off];
    }
    __syncthreads();
  }
  const float invp = rsqrtf(fmaxf(red[0][0], 1e-12f));
  const float invq = rsqrtf(fmaxf(red[1][0], 1e-12f));
#pragma unroll
  for (int i = 0; i < 3; ++i) {
    const int d = tid + i * 256;
    pn[(long)row * D_ + d] = (_Float16)(pv[i] * invp);
    qn[(long)row * D_ + d] = (_Float16)(qv[i] * invq);
    // raw q, transposed per batch: qT[b][d][s]
    qT[(long)b * D_ * S_ + (long)d * S_ + s] = (_Float16)qv[i];
  }
}

// ---------------------------------------------------------------------------
// Batched GEMM: C[b] = scale * A[b] (MxK, row-major) x Bt[b]^T (Bt is NxK
// row-major). 128x256 tile per block, 8 waves in a 2x4 grid, each wave a
// 64x64 sub-tile via 4x4 16x16x32 f16 WMMA fragments (16 wmma : 16 ds_load
// per step). Double-buffered LDS fed by async LDS-direct copies (ASYNCcnt).
// ---------------------------------------------------------------------------
template <bool OUT_HALF>
__global__ __launch_bounds__(256) void gemm_tt(
    const _Float16* __restrict__ A, const _Float16* __restrict__ Bt,
    void* __restrict__ C, int M, int N, int K,
    long sA_, long sB_, long sC_, float scale)
{
  __shared__ _Float16 smA[2][BM * LDT];   // 2 x 5120 halfs
  __shared__ _Float16 smB[2][BN * LDT];   // 2 x 10240 halfs

  const int bz = blockIdx.z;
  const int tm = blockIdx.y * BM;
  const int tn = blockIdx.x * BN;
  const _Float16* Ab = A  + bz * sA_ + (long)tm * K;
  const _Float16* Bb = Bt + bz * sB_ + (long)tn * K;

  const int tid  = threadIdx.x;
  const int lane = tid & 31, wid = tid >> 5;
  const int wm = wid >> 2, wn = wid & 3;     // 2 x 4 wave grid -> 64x64 tiles
  const int hi = lane >> 4, lr = lane & 15;

  v8f acc[4][4];
  {
    v8f zero = {};
#pragma unroll
    for (int i = 0; i < 4; ++i)
#pragma unroll
      for (int j = 0; j < 4; ++j) acc[i][j] = zero;
  }

  // ---- staging helper offsets: A = 512 16B-quads, B = 1024 16B-quads ----
  // prologue: stage k-block 0 into buffer 0 (async LDS-direct copies)
#pragma unroll
  for (int r = 0; r < 2; ++r) {
    const int qd = tid + r * 256;
    const int row = qd >> 2, c4 = qd & 3;
    copy16_g2l(Ab + (long)row * K + c4 * 8, &smA[0][row * LDT + c4 * 8]);
  }
#pragma unroll
  for (int r = 0; r < 4; ++r) {
    const int qd = tid + r * 256;
    const int row = qd >> 2, c4 = qd & 3;
    copy16_g2l(Bb + (long)row * K + c4 * 8, &smB[0][row * LDT + c4 * 8]);
  }
  async_drain();
  __syncthreads();

  const int nsteps = K / BK;
  for (int step = 0; step < nsteps; ++step) {
    const int buf = step & 1;
    if (step + 1 < nsteps) {
      const int kk = (step + 1) * BK;
      _Float16* dA = &smA[buf ^ 1][0];
      _Float16* dB = &smB[buf ^ 1][0];
#pragma unroll
      for (int r = 0; r < 2; ++r) {
        const int qd = tid + r * 256;
        const int row = qd >> 2, c4 = qd & 3;
        copy16_g2l(Ab + (long)row * K + kk + c4 * 8, dA + row * LDT + c4 * 8);
      }
#pragma unroll
      for (int r = 0; r < 4; ++r) {
        const int qd = tid + r * 256;
        const int row = qd >> 2, c4 = qd & 3;
        copy16_g2l(Bb + (long)row * K + kk + c4 * 8, dB + row * LDT + c4 * 8);
      }
    }

    union Frag { v16h h; unsigned u[8]; };
    Frag af[4], bf[4];
#pragma unroll
    for (int fm = 0; fm < 4; ++fm) {
      const int rowb = (wm * 64 + fm * 16 + lr) * LDT;
#pragma unroll
      for (int i = 0; i < 8; ++i) {
        const int kb = (i & 3) * 2 + hi * 8 + (i >> 2) * 16;  // 16-bit A layout
        af[fm].u[i] = *(const unsigned*)&smA[buf][rowb + kb];
      }
    }
#pragma unroll
    for (int fn = 0; fn < 4; ++fn) {
      const int rowb = (wn * 64 + fn * 16 + lr) * LDT;
#pragma unroll
      for (int i = 0; i < 8; ++i) {
        const int kb = (i & 3) * 2 + hi * 8 + (i >> 2) * 16;
        bf[fn].u[i] = *(const unsigned*)&smB[buf][rowb + kb];
      }
    }

#pragma unroll
    for (int fm = 0; fm < 4; ++fm)
#pragma unroll
      for (int fn = 0; fn < 4; ++fn)
        acc[fm][fn] = __builtin_amdgcn_wmma_f32_16x16x32_f16(
            false, af[fm].h, false, bf[fn].h, (short)0, acc[fm][fn],
            false, false);

    async_drain();     // my async copies into buf^1 have landed in LDS
    __syncthreads();   // everyone's copies visible; everyone done reading buf
  }

  // epilogue: C/D layout -> lane l, vgpr v: row = v + 8*(l/16), col = l%16
#pragma unroll
  for (int fm = 0; fm < 4; ++fm) {
#pragma unroll
    for (int fn = 0; fn < 4; ++fn) {
      const int row0 = tm + wm * 64 + fm * 16 + hi * 8;
      const int col  = tn + wn * 64 + fn * 16 + lr;
#pragma unroll
      for (int v = 0; v < 8; ++v) {
        const float val = acc[fm][fn][v] * scale;
        const long idx = bz * sC_ + (long)(row0 + v) * N + col;
        if (OUT_HALF) ((_Float16*)C)[idx] = (_Float16)val;
        else          ((float*)C)[idx]    = val;
      }
    }
  }
}

// ---------------------------------------------------------------------------
// Kernel 4: word_dist[b,s] = -(1/D) * dot(normalize(p_row), normalize(mean_row))
// ---------------------------------------------------------------------------
__global__ __launch_bounds__(256) void finalize_kernel(
    const float* __restrict__ p, const float* __restrict__ mean,
    float* __restrict__ out)
{
  const int row = blockIdx.x;
  const int tid = threadIdx.x;
  const float* pr = p + (long)row * D_;
  const float* mr = mean + (long)row * D_;

  float ssp = 0.f, ssm = 0.f, dot = 0.f;
#pragma unroll
  for (int i = 0; i < 3; ++i) {
    const float x = pr[tid + i * 256];
    const float m = mr[tid + i * 256];
    ssp += x * x; ssm += m * m; dot += x * m;
  }
  __shared__ float red[3][256];
  red[0][tid] = ssp; red[1][tid] = ssm; red[2][tid] = dot;
  __syncthreads();
  for (int off = 128; off > 0; off >>= 1) {
    if (tid < off) {
      red[0][tid] += red[0][tid + off];
      red[1][tid] += red[1][tid + off];
      red[2][tid] += red[2][tid + off];
    }
    __syncthreads();
  }
  if (tid == 0) {
    const float invp = rsqrtf(fmaxf(red[0][0], 1e-12f));
    const float invm = rsqrtf(fmaxf(red[1][0], 1e-12f));
    out[row] = -(red[2][0] * invp * invm) * (1.0f / (float)D_);
  }
}

// ---------------------------------------------------------------------------
extern "C" void kernel_launch(void* const* d_in, const int* in_sizes, int n_in,
                              void* d_out, int out_size, void* d_ws, size_t ws_size,
                              hipStream_t stream)
{
  const float* p = (const float*)d_in[0];
  const float* q = (const float*)d_in[1];

  // workspace layout (bytes)
  char* ws = (char*)d_ws;
  _Float16* pn   = (_Float16*)(ws + 0);           //  48 MB: [B][S][D] f16
  _Float16* qn   = (_Float16*)(ws + 50331648);    //  48 MB: [B][S][D] f16
  _Float16* qT   = (_Float16*)(ws + 100663296);   //  48 MB: [B][D][S] f16 (raw q)
  _Float16* w    = (_Float16*)(ws + 150994944);   //  32 MB: [B][S][S] f16 (= -cos)
  float*    mean = (float*)   (ws + 184549376);   //  96 MB: [B][S][D] f32

  prep_kernel<<<B_ * S_, 256, 0, stream>>>(p, q, pn, qn, qT);

  // w = -(pn . qn^T)   (1/D folded into GEMM2 instead, keeps w in f16 range)
  gemm_tt<true><<<dim3(S_ / BN, S_ / BM, B_), 256, 0, stream>>>(
      pn, qn, (void*)w, S_, S_, D_,
      (long)S_ * D_, (long)S_ * D_, (long)S_ * S_, -1.0f);

  // mean = (1/(D*S)) * (w . q)   via qT as NxK operand
  gemm_tt<false><<<dim3(D_ / BN, S_ / BM, B_), 256, 0, stream>>>(
      w, qT, (void*)mean, S_, D_, S_,
      (long)S_ * S_, (long)D_ * S_, (long)S_ * D_,
      1.0f / ((float)D_ * (float)S_));

  finalize_kernel<<<B_ * S_, 256, 0, stream>>>(p, mean, (float*)d_out);
}
